// Reliable_Transformer_37125697306733
// MI455X (gfx1250) — matrix operand order repaired and checked by reading, not compiled, not run on hardware
//
#include <hip/hip_runtime.h>
#include <hip/hip_bf16.h>
#include <math.h>

// ---------------------------------------------------------------------------
// CDNA5 / gfx1250. Heavy GEMMs + convs use v_wmma_f32_16x16x32_bf16.
// Activations that feed GEMMs are stored bf16 by their producers, so A
// fragments are two b128 loads + register concat (no converts in the loop).
// Weights are pre-packed transposed bf16 -> B fragment is one 32B load.
// OOB handled by address clamping (garbage rows/cols never stored).
// ---------------------------------------------------------------------------

typedef __bf16 bf16_t;
typedef __attribute__((ext_vector_type(16))) __bf16 v16bf;
typedef __attribute__((ext_vector_type(8)))  __bf16 v8bf;
typedef __attribute__((ext_vector_type(8)))  float  v8f;

static __device__ __forceinline__ float gelu_f(float x) {
    const float k0 = 0.7978845608028654f;  // sqrt(2/pi)
    float x3 = x * x * x;
    return 0.5f * x * (1.f + tanhf(k0 * (x + 0.044715f * x3)));
}

static __device__ __forceinline__ v16bf cat16(v8bf lo, v8bf hi) {
    return __builtin_shufflevector(lo, hi, 0, 1, 2, 3, 4, 5, 6, 7,
                                   8, 9, 10, 11, 12, 13, 14, 15);
}

// ============ weight packing: Wt[l][n][k] = bf16(W[l][k][n]) ===============
__global__ void pack_wt_kernel(const float* __restrict__ W, bf16_t* __restrict__ Wt,
                               int K, int N, int L) {
    long g = (long)blockIdx.x * blockDim.x + threadIdx.x;
    long total = (long)L * N * K;
    if (g >= total) return;
    int k = (int)(g % K);
    long t = g / K;
    int n = (int)(t % N);
    int l = (int)(t / N);
    Wt[g] = (bf16_t)W[((size_t)l * K + k) * N + n];
}

__global__ void cvt_bf16_kernel(const float* __restrict__ W, bf16_t* __restrict__ Wt,
                                long n) {
    long g = (long)blockIdx.x * blockDim.x + threadIdx.x;
    if (g >= n) return;
    Wt[g] = (bf16_t)W[g];
}

// ======================= WMMA GEMM (bf16 in, f32 acc) ======================
// C[M,N] = act(A[M,K] @ B[K,N] + bias) + res. A: bf16 [M][K]; Bt: bf16 [N][K].
// One wave computes a 64x16 tile (4 M-subtiles sharing one B fragment).
// Exactly one of Cf / Cb is non-null (f32 or bf16 output).
#define MT 4
__global__ void gemm_bf16_wmma(const bf16_t* __restrict__ A, int lda,
                               const bf16_t* __restrict__ Bt,
                               const float* __restrict__ bias,
                               const float* __restrict__ res, int ldr,
                               float* __restrict__ Cf, bf16_t* __restrict__ Cb,
                               int ldc, int M, int N, int K, int act) {
    int wave = threadIdx.x >> 5;
    int lane = threadIdx.x & 31;
    int tilesN = N >> 4;                      // N is always a multiple of 16
    long tilesM = ((long)M + 63) >> 6;        // 64 rows per wave
    long tile = (long)blockIdx.x * (blockDim.x >> 5) + wave;
    if (tile >= tilesM * tilesN) return;      // wave-uniform exit
    int tn = (int)(tile % tilesN);
    long tm = tile / tilesN;
    int l15 = lane & 15;
    int half = lane >> 4;
    int ncol = tn * 16 + l15;
    const bf16_t* Brow = Bt + (size_t)ncol * K + 16 * half;
    const bf16_t* Ar[MT];
#pragma unroll
    for (int i = 0; i < MT; ++i) {
        long mr = tm * 64 + i * 16 + l15;
        if (mr >= M) mr = M - 1;              // clamp: garbage rows never stored
        Ar[i] = A + (size_t)mr * lda + 16 * half;
    }
    v8f z = {0.f, 0.f, 0.f, 0.f, 0.f, 0.f, 0.f, 0.f};
    v8f acc[MT] = {z, z, z, z};
    for (int kt = 0; kt < K; kt += 32) {
        // hoist all loads (independent dests -> one clause, one wait)
        v16bf b = *(const v16bf*)(Brow + kt);
        v8bf lo[MT], hi[MT];
#pragma unroll
        for (int i = 0; i < MT; ++i) {
            const bf16_t* p = Ar[i] + kt;
            lo[i] = *(const v8bf*)(p);
            hi[i] = *(const v8bf*)(p + 8);
        }
#pragma unroll
        for (int i = 0; i < MT; ++i) {
            v16bf a = cat16(lo[i], hi[i]);    // register concat, no ALU
            acc[i] = __builtin_amdgcn_wmma_f32_16x16x32_bf16(false, a, false, b,
                                                             (short)0, acc[i], false, false);
        }
    }
    float bs = bias ? bias[ncol] : 0.f;
#pragma unroll
    for (int i = 0; i < MT; ++i) {
#pragma unroll
        for (int j = 0; j < 8; ++j) {
            long mr = tm * 64 + i * 16 + half * 8 + j;
            if (mr < M) {
                float v = acc[i][j] + bs;
                if (act == 1) v = gelu_f(v);
                if (res) v += res[(size_t)mr * ldr + ncol];
                if (Cb) Cb[(size_t)mr * ldc + ncol] = (bf16_t)v;
                else    Cf[(size_t)mr * ldc + ncol] = v;
            }
        }
    }
}

// ================= 3x3 conv as implicit GEMM (WMMA, bf16) ==================
// X: bf16 [B,inC,H,W]. Wt: bf16 [outC][inC*9] (OIHW is already im2col order).
// Loads unconditional (clamped addresses); border zeroing via value selects.
__global__ void conv3x3_wmma(const bf16_t* __restrict__ X,
                             const bf16_t* __restrict__ Wt,
                             const float* __restrict__ bias,
                             const float* __restrict__ bng,
                             const float* __restrict__ bnb,
                             float* __restrict__ Yf, bf16_t* __restrict__ Yb,
                             int Bb, int inC, int outC, int H, int Wd, int relu) {
    int wave = threadIdx.x >> 5;
    int lane = threadIdx.x & 31;
    int K = inC * 9;
    long M = (long)Bb * H * Wd;
    int tilesN = (outC + 15) >> 4;
    long tilesM = (M + 15) >> 4;
    long tile = (long)blockIdx.x * (blockDim.x >> 5) + wave;
    if (tile >= tilesM * tilesN) return;
    int tn = (int)(tile % tilesN);
    long tm = tile / tilesN;
    int l15 = lane & 15;
    int half = lane >> 4;
    long mrow = tm * 16 + l15;
    long mrowc = (mrow < M) ? mrow : M - 1;
    int ncol = tn * 16 + l15;
    int ncolc = (ncol < outC) ? ncol : outC - 1;
    int HW = H * Wd;
    int bi = (int)(mrowc / HW);
    int pix = (int)(mrowc % HW);
    int py = pix / Wd, px = pix % Wd;
    const bf16_t* Xb = X + (size_t)bi * inC * HW;
    const bf16_t* Brow = Wt + (size_t)ncolc * K + 16 * half;
    v8f acc = {0.f, 0.f, 0.f, 0.f, 0.f, 0.f, 0.f, 0.f};
    const bf16_t zb = (bf16_t)0.f;
    for (int kt = 0; kt < K; kt += 32) {
        v16bf b = *(const v16bf*)(Brow + kt);
        bf16_t raw[16];
        bool inb[16];
#pragma unroll
        for (int j = 0; j < 16; ++j) {
            int ka = kt + j + 16 * half;
            int ci = ka / 9, rem = ka % 9;
            int sy = py + rem / 3 - 1;
            int sx = px + rem % 3 - 1;
            int syc = min(max(sy, 0), H - 1);
            int sxc = min(max(sx, 0), Wd - 1);
            inb[j] = (sy == syc) && (sx == sxc);
            raw[j] = Xb[((size_t)ci * H + syc) * Wd + sxc];
        }
        v16bf a;
#pragma unroll
        for (int j = 0; j < 16; ++j) a[j] = inb[j] ? raw[j] : zb;
        acc = __builtin_amdgcn_wmma_f32_16x16x32_bf16(false, a, false, b,
                                                      (short)0, acc, false, false);
    }
    if (ncol < outC) {
#pragma unroll
        for (int j = 0; j < 8; ++j) {
            long mr = tm * 16 + half * 8 + j;
            if (mr < M) {
                int bi2 = (int)(mr / HW), pix2 = (int)(mr % HW);
                float v = acc[j];
                if (bias) v += bias[ncol];
                if (bng)  v = v * bng[ncol] + bnb[ncol];
                if (relu) v = fmaxf(v, 0.f);
                size_t o = ((size_t)bi2 * outC + ncol) * HW + pix2;
                if (Yb) Yb[o] = (bf16_t)v;
                else    Yf[o] = v;
            }
        }
    }
}

// ============================ small kernels ================================
__global__ void absdiff_rows_kernel(const float* __restrict__ x1,
                                    const float* __restrict__ x2,
                                    float* __restrict__ af) {
    long g = (long)blockIdx.x * blockDim.x + threadIdx.x;
    if (g >= (long)16384 * 32) return;
    int c = (int)(g & 31);
    long row = g >> 5;
    int bi = (int)(row >> 12);
    int p = (int)(row & 4095);
    size_t src = ((size_t)bi * 32 + c) * 4096 + p;
    af[g] = fabsf(x1[src] - x2[src]);
}

__global__ void edge_dot_kernel(const float* __restrict__ af,
                                const int* __restrict__ ind0,
                                float* __restrict__ vals, int E) {
    int e = blockIdx.x * blockDim.x + threadIdx.x;
    if (e >= E) return;
    const float* a = af + (size_t)ind0[e] * 32;
    const float* b = af + (size_t)ind0[E + e] * 32;
    float s = 0.f;
#pragma unroll
    for (int c = 0; c < 32; ++c) s += a[c] * b[c];
    vals[e] = s;
}

__global__ void dis_kernel(const float* __restrict__ vals, float* __restrict__ dis) {
    int i = blockIdx.x * blockDim.x + threadIdx.x;
    if (i >= 16384) return;
    float d = 1.f;  // +identity
#pragma unroll
    for (int k = 0; k < 8; ++k) d += vals[i * 8 + k];
    dis[i] = rsqrtf(fmaxf(d, 1e-6f));
}

__global__ void score_kernel(const float* __restrict__ af,
                             const float* __restrict__ gcw,
                             float* __restrict__ sg) {
    int i = blockIdx.x * blockDim.x + threadIdx.x;
    if (i >= 16384) return;
    float s = 0.f;
#pragma unroll
    for (int c = 0; c < 32; ++c) s += af[(size_t)i * 32 + c] * gcw[c];
    sg[i] = s;
}

__global__ void fm_kernel(const float* __restrict__ vals,
                          const int* __restrict__ ind0,
                          const float* __restrict__ dis,
                          const float* __restrict__ sg,
                          const float* __restrict__ gcb,
                          float* __restrict__ fm, int E) {
    int i = blockIdx.x * blockDim.x + threadIdx.x;
    if (i >= 16384) return;
    float acc = dis[i] * sg[i];  // identity diagonal term
#pragma unroll
    for (int k = 0; k < 8; ++k) {
        int e = i * 8 + k;
        int col = ind0[E + e];   // global flat neighbor index
        acc += vals[e] * dis[col] * sg[col];
    }
    fm[i] = dis[i] * acc + gcb[0];
}

// Exact stable top-k-smallest via ranking (matches lax.top_k(-x) order).
__global__ void topk_rank_kernel(const float* __restrict__ fm, int* __restrict__ idx) {
    int bi = blockIdx.y;
    int j = blockIdx.x * blockDim.x + threadIdx.x;
    if (j >= 4096) return;
    const float* f = fm + bi * 4096;
    float v = f[j];
    int rank = 0;
    for (int t = 0; t < 4096; ++t) {
        float u = f[t];
        rank += (u < v) || (u == v && t < j);
    }
    if (rank < 1024) idx[bi * 1024 + rank] = j;
}

__global__ void gather_tok_kernel(const float* __restrict__ X,
                                  const int* __restrict__ idx,
                                  float* __restrict__ out) {
    long g = (long)blockIdx.x * blockDim.x + threadIdx.x;
    if (g >= (long)4 * 1024 * 32) return;
    int c = (int)(g & 31);
    int t = (int)((g >> 5) & 1023);
    int bi = (int)(g >> 15);
    int p = idx[bi * 1024 + t];
    out[g] = X[((size_t)bi * 32 + c) * 4096 + p];
}

// K-means: 10 scan iterations + final scatter-mean. One block per batch.
__global__ void kmeans_kernel(const float* __restrict__ tok, float* __restrict__ outc) {
    int bi = blockIdx.x;
    const float* T = tok + (size_t)bi * 1024 * 32;
    __shared__ float cent[10][32];
    __shared__ float accum[10][32];
    __shared__ float cnt[10];
    int tid = threadIdx.x;
    for (int i = tid; i < 10 * 32; i += blockDim.x) cent[i / 32][i % 32] = T[i];
    __syncthreads();
    for (int iter = 0; iter <= 10; ++iter) {
        for (int i = tid; i < 10 * 32; i += blockDim.x) accum[i / 32][i % 32] = 0.f;
        if (tid < 10) cnt[tid] = 0.f;
        __syncthreads();
        for (int t = tid; t < 1024; t += blockDim.x) {
            const float* x = T + t * 32;
            int best = 0;
            float bd = 3.4e38f;
            for (int k = 0; k < 10; ++k) {
                float d = 0.f;
#pragma unroll
                for (int c = 0; c < 32; ++c) {
                    float df = x[c] - cent[k][c];
                    d += df * df;
                }
                if (d < bd) { bd = d; best = k; }
            }
#pragma unroll
            for (int c = 0; c < 32; ++c) atomicAdd(&accum[best][c], x[c]);
            atomicAdd(&cnt[best], 1.f);
        }
        __syncthreads();
        if (iter < 10) {
            for (int i = tid; i < 10 * 32; i += blockDim.x) {
                int k = i / 32, c = i % 32;
                float cn = cnt[k];
                float nw = accum[k][c] / fmaxf(cn, 1.f);
                cent[k][c] = (cn > 0.f) ? nw : cent[k][c];
            }
            __syncthreads();
        } else {
            for (int i = tid; i < 10 * 32; i += blockDim.x) {
                int k = i / 32;
                outc[(size_t)bi * 320 + i] = accum[k][i % 32] / fmaxf(cnt[k], 1.f);
            }
        }
    }
}

__global__ void concat_pos_kernel(const float* __restrict__ t1,
                                  const float* __restrict__ t2,
                                  const float* __restrict__ pos,
                                  float* __restrict__ tokens) {
    int g = blockIdx.x * blockDim.x + threadIdx.x;
    if (g >= 4 * 20 * 32) return;
    int c = g & 31;
    int t = (g >> 5) % 20;
    int bi = g / (20 * 32);
    float v = (t < 10) ? t1[(bi * 10 + t) * 32 + c] : t2[(bi * 10 + t - 10) * 32 + c];
    tokens[g] = v + pos[t * 32 + c];
}

__global__ void split_tokens_kernel(const float* __restrict__ tokens,
                                    float* __restrict__ t1, float* __restrict__ t2) {
    int g = blockIdx.x * blockDim.x + threadIdx.x;
    if (g >= 4 * 20 * 32) return;
    int c = g & 31;
    int t = (g >> 5) % 20;
    int bi = g / (20 * 32);
    if (t < 10) t1[(bi * 10 + t) * 32 + c] = tokens[g];
    else        t2[(bi * 10 + (t - 10)) * 32 + c] = tokens[g];
}

// LayerNorm over C=32; output bf16 (feeds GEMM A operands only).
__global__ void ln_kernel(const float* __restrict__ x, const float* __restrict__ g,
                          const float* __restrict__ b, bf16_t* __restrict__ y, int M) {
    int r = blockIdx.x * blockDim.x + threadIdx.x;
    if (r >= M) return;
    const float* xr = x + (size_t)r * 32;
    float mu = 0.f;
#pragma unroll
    for (int c = 0; c < 32; ++c) mu += xr[c];
    mu *= (1.f / 32.f);
    float va = 0.f;
#pragma unroll
    for (int c = 0; c < 32; ++c) {
        float d = xr[c] - mu;
        va += d * d;
    }
    va *= (1.f / 32.f);
    float inv = rsqrtf(va + 1e-5f);
    bf16_t* yr = y + (size_t)r * 32;
#pragma unroll
    for (int c = 0; c < 32; ++c) yr[c] = (bf16_t)((xr[c] - mu) * inv * g[c] + b[c]);
}

// Fused MHA (8 heads, dh=64) against NKEYS keys; thread per (row, head).
// Output bf16 (feeds out-projection GEMM A operand).
template <int NKEYS>
__global__ void attn_kernel(const float* __restrict__ q, int ldq,
                            const float* __restrict__ k, int ldk,
                            const float* __restrict__ v, int ldv,
                            bf16_t* __restrict__ out, int ldo,
                            int rows_per_batch, int total_rows) {
    long gid = (long)blockIdx.x * blockDim.x + threadIdx.x;
    long row = gid >> 3;
    int head = (int)(gid & 7);
    if (row >= total_rows) return;
    int batch = (int)(row / rows_per_batch);
    const float* qr = q + (size_t)row * ldq + head * 64;
    float qv[64];
#pragma unroll
    for (int t = 0; t < 64; ++t) qv[t] = qr[t];
    float logit[NKEYS];
    float mx = -3.4e38f;
#pragma unroll
    for (int j = 0; j < NKEYS; ++j) {
        const float* kr = k + (size_t)(batch * NKEYS + j) * ldk + head * 64;
        float d = 0.f;
#pragma unroll
        for (int t = 0; t < 64; ++t) d += qv[t] * kr[t];
        d *= 0.125f;  // 1/sqrt(64)
        logit[j] = d;
        mx = fmaxf(mx, d);
    }
    float se = 0.f;
#pragma unroll
    for (int j = 0; j < NKEYS; ++j) {
        float e = __expf(logit[j] - mx);
        logit[j] = e;
        se += e;
    }
    float inv = 1.f / se;
    float o[64];
#pragma unroll
    for (int t = 0; t < 64; ++t) o[t] = 0.f;
#pragma unroll
    for (int j = 0; j < NKEYS; ++j) {
        float w = logit[j] * inv;
        const float* vr = v + (size_t)(batch * NKEYS + j) * ldv + head * 64;
#pragma unroll
        for (int t = 0; t < 64; ++t) o[t] += w * vr[t];
    }
    bf16_t* orow = out + (size_t)row * ldo + head * 64;
#pragma unroll
    for (int t = 0; t < 64; ++t) orow[t] = (bf16_t)o[t];
}

__global__ void nchw_to_rows_kernel(const float* __restrict__ X, float* __restrict__ t) {
    long g = (long)blockIdx.x * blockDim.x + threadIdx.x;
    if (g >= (long)16384 * 32) return;
    int c = (int)(g & 31);
    long row = g >> 5;
    int bi = (int)(row >> 12);
    int p = (int)(row & 4095);
    t[g] = X[((size_t)bi * 32 + c) * 4096 + p];
}

__global__ void diff_to_nchw_kernel(const float* __restrict__ d1,
                                    const float* __restrict__ d2,
                                    float* __restrict__ out) {
    long g = (long)blockIdx.x * blockDim.x + threadIdx.x;
    if (g >= (long)4 * 32 * 4096) return;
    int p = (int)(g & 4095);
    int c = (int)((g >> 12) & 31);
    int bi = (int)(g >> 17);
    size_t r = ((size_t)bi * 4096 + p) * 32 + c;
    out[g] = fabsf(d1[r] - d2[r]);
}

// Bilinear x4 upsample; output bf16 (feeds conv1 A operand).
__global__ void upsample4_kernel(const float* __restrict__ X, bf16_t* __restrict__ Y) {
    long g = (long)blockIdx.x * blockDim.x + threadIdx.x;
    if (g >= (long)4 * 32 * 256 * 256) return;
    int x = (int)(g & 255);
    int y = (int)((g >> 8) & 255);
    int c = (int)((g >> 16) & 31);
    int bi = (int)(g >> 21);
    float sy = (y + 0.5f) * 0.25f - 0.5f;
    float sx = (x + 0.5f) * 0.25f - 0.5f;
    int y0 = (int)floorf(sy), x0 = (int)floorf(sx);
    float fy = sy - y0, fx = sx - x0;
    int y0c = min(max(y0, 0), 63), y1c = min(max(y0 + 1, 0), 63);
    int x0c = min(max(x0, 0), 63), x1c = min(max(x0 + 1, 0), 63);
    const float* P = X + ((size_t)bi * 32 + c) * 4096;
    float v = (1.f - fy) * ((1.f - fx) * P[y0c * 64 + x0c] + fx * P[y0c * 64 + x1c]) +
              fy * ((1.f - fx) * P[y1c * 64 + x0c] + fx * P[y1c * 64 + x1c]);
    Y[g] = (bf16_t)v;
}

// ============================== host side ==================================
// Input order assumption: top-level setup_inputs() insertion order
// (x1, x2, indice0, indice1), then params flattened with JAX pytree
// convention (dict keys sorted at every level) -> 49 leaves total.
enum {
    IN_X1 = 0, IN_X2, IN_IND0, IN_IND1,
    P_BN_B, P_BN_G, P_CLS_B2, P_CLS_W1, P_CLS_W2,
    PC_FF_B1, PC_FF_B2, PC_FF_W1, PC_FF_W2, PC_K_W, PC_LN1_B, PC_LN1_G,
    PC_LN2_B, PC_LN2_G, PC_OUT_B, PC_OUT_W, PC_Q_W, PC_V_W,
    PD_FF_B1, PD_FF_B2, PD_FF_W1, PD_FF_W2, PD_K_W, PD_LN1_B, PD_LN1_G,
    PD_LN2_B, PD_LN2_G, PD_OUT_B, PD_OUT_W, PD_Q_W, PD_V_W,
    PE_FF_B1, PE_FF_B2, PE_FF_W1, PE_FF_W2, PE_LN1_B, PE_LN1_G,
    PE_LN2_B, PE_LN2_G, PE_OUT_B, PE_OUT_W, PE_QKV_W,
    P_GC_B, P_GC_W, P_POS, IN_COUNT
};

extern "C" void kernel_launch(void* const* d_in, const int* in_sizes, int n_in,
                              void* d_out, int out_size, void* d_ws, size_t ws_size,
                              hipStream_t stream) {
    (void)out_size; (void)ws_size;
    if (n_in < IN_COUNT) return;
    constexpr int BB = 4, CC = 32, NPIX = 4096, RWS = 16384, KTOP = 1024;
    constexpr int CLUST = 10, INR = 512, MLPD = 64, TT = 20, DEPTH = 8;
    auto F = [&](int i) { return (const float*)d_in[i]; };
    const int* ind0 = (const int*)d_in[IN_IND0];
    const int E = in_sizes[IN_IND0] / 2;

    // ---- bump allocator over d_ws ----
    char* base = (char*)d_ws;
    size_t off = 0;
    auto alloc = [&](size_t bytes) -> float* {
        off = (off + 255) & ~(size_t)255;
        float* p = (float*)(base + off);
        off += bytes;
        return p;
    };
    float* af   = alloc((size_t)RWS * CC * 4);
    float* vals = alloc((size_t)E * 4);
    float* dis  = alloc((size_t)RWS * 4);
    float* sg   = alloc((size_t)RWS * 4);
    float* fmv  = alloc((size_t)RWS * 4);
    int*   idx  = (int*)alloc((size_t)BB * KTOP * 4);
    float* tg1  = alloc((size_t)BB * KTOP * CC * 4);
    float* tg2  = alloc((size_t)BB * KTOP * CC * 4);
    float* tok1 = alloc((size_t)BB * CLUST * CC * 4);
    float* tok2 = alloc((size_t)BB * CLUST * CC * 4);
    float* tokens  = alloc((size_t)BB * TT * CC * 4);
    float* tokens2 = alloc((size_t)BB * TT * CC * 4);
    bf16_t* xn_tok  = (bf16_t*)alloc((size_t)BB * TT * CC * 2);
    float* qkv_tok = alloc((size_t)BB * TT * 3 * INR * 4);
    bf16_t* attn_tok = (bf16_t*)alloc((size_t)BB * TT * INR * 2);
    bf16_t* h_tok   = (bf16_t*)alloc((size_t)BB * TT * MLPD * 2);
    float* t1e = alloc((size_t)BB * CLUST * CC * 4);
    float* t2e = alloc((size_t)BB * CLUST * CC * 4);
    float* t1c = alloc((size_t)BB * CLUST * CC * 4);
    float* t2c = alloc((size_t)BB * CLUST * CC * 4);
    bf16_t* xn40 = (bf16_t*)alloc((size_t)BB * CLUST * CC * 2);
    bf16_t* mn40 = (bf16_t*)alloc((size_t)BB * CLUST * CC * 2);
    float* q40 = alloc((size_t)BB * CLUST * INR * 4);
    float* k40 = alloc((size_t)BB * CLUST * INR * 4);
    float* v40 = alloc((size_t)BB * CLUST * INR * 4);
    bf16_t* ao40 = (bf16_t*)alloc((size_t)BB * CLUST * INR * 2);
    bf16_t* h40  = (bf16_t*)alloc((size_t)BB * CLUST * MLPD * 2);
    float* tcur = alloc((size_t)RWS * CC * 4);
    bf16_t* xnb  = (bf16_t*)alloc((size_t)RWS * CC * 2);
    bf16_t* mnb  = (bf16_t*)alloc((size_t)BB * CLUST * CC * 2);
    float* qb   = alloc((size_t)RWS * INR * 4);   // 32MB
    float* kb   = alloc((size_t)BB * CLUST * INR * 4);
    float* vb   = alloc((size_t)BB * CLUST * INR * 4);
    bf16_t* aob = (bf16_t*)alloc((size_t)RWS * INR * 2);
    bf16_t* hb  = (bf16_t*)alloc((size_t)RWS * MLPD * 2);
    float* dec1 = alloc((size_t)RWS * CC * 4);
    float* dec2 = alloc((size_t)RWS * CC * 4);
    float* diffn = alloc((size_t)RWS * CC * 4);
    bf16_t* ups = (bf16_t*)alloc((size_t)BB * CC * 256 * 256 * 2);
    bf16_t* c1o = (bf16_t*)alloc((size_t)BB * CC * 256 * 256 * 2);
    // packed bf16 weights
    bf16_t* pe_qkv_t = (bf16_t*)alloc((size_t)CC * 3 * INR * 2);
    bf16_t* pe_out_t = (bf16_t*)alloc((size_t)INR * CC * 2);
    bf16_t* pe_ff1_t = (bf16_t*)alloc((size_t)CC * MLPD * 2);
    bf16_t* pe_ff2_t = (bf16_t*)alloc((size_t)MLPD * CC * 2);
    bf16_t* pc_q_t = (bf16_t*)alloc((size_t)CC * INR * 2);
    bf16_t* pc_k_t = (bf16_t*)alloc((size_t)CC * INR * 2);
    bf16_t* pc_v_t = (bf16_t*)alloc((size_t)CC * INR * 2);
    bf16_t* pc_out_t = (bf16_t*)alloc((size_t)INR * CC * 2);
    bf16_t* pc_ff1_t = (bf16_t*)alloc((size_t)CC * MLPD * 2);
    bf16_t* pc_ff2_t = (bf16_t*)alloc((size_t)MLPD * CC * 2);
    bf16_t* pd_q_t = (bf16_t*)alloc((size_t)DEPTH * CC * INR * 2);
    bf16_t* pd_k_t = (bf16_t*)alloc((size_t)DEPTH * CC * INR * 2);
    bf16_t* pd_v_t = (bf16_t*)alloc((size_t)DEPTH * CC * INR * 2);
    bf16_t* pd_out_t = (bf16_t*)alloc((size_t)DEPTH * INR * CC * 2);
    bf16_t* pd_ff1_t = (bf16_t*)alloc((size_t)DEPTH * CC * MLPD * 2);
    bf16_t* pd_ff2_t = (bf16_t*)alloc((size_t)DEPTH * MLPD * CC * 2);
    bf16_t* w1c = (bf16_t*)alloc((size_t)CC * CC * 9 * 2);
    bf16_t* w2c = (bf16_t*)alloc((size_t)2 * CC * 9 * 2);

    auto pack = [&](const float* W, bf16_t* Wt, int K, int N, int L) {
        long total = (long)L * N * K;
        pack_wt_kernel<<<(int)((total + 255) / 256), 256, 0, stream>>>(W, Wt, K, N, L);
    };
    pack(F(PE_QKV_W), pe_qkv_t, CC, 3 * INR, 1);
    pack(F(PE_OUT_W), pe_out_t, INR, CC, 1);
    pack(F(PE_FF_W1), pe_ff1_t, CC, MLPD, 1);
    pack(F(PE_FF_W2), pe_ff2_t, MLPD, CC, 1);
    pack(F(PC_Q_W), pc_q_t, CC, INR, 1);
    pack(F(PC_K_W), pc_k_t, CC, INR, 1);
    pack(F(PC_V_W), pc_v_t, CC, INR, 1);
    pack(F(PC_OUT_W), pc_out_t, INR, CC, 1);
    pack(F(PC_FF_W1), pc_ff1_t, CC, MLPD, 1);
    pack(F(PC_FF_W2), pc_ff2_t, MLPD, CC, 1);
    pack(F(PD_Q_W), pd_q_t, CC, INR, DEPTH);
    pack(F(PD_K_W), pd_k_t, CC, INR, DEPTH);
    pack(F(PD_V_W), pd_v_t, CC, INR, DEPTH);
    pack(F(PD_OUT_W), pd_out_t, INR, CC, DEPTH);
    pack(F(PD_FF_W1), pd_ff1_t, CC, MLPD, DEPTH);
    pack(F(PD_FF_W2), pd_ff2_t, MLPD, CC, DEPTH);
    cvt_bf16_kernel<<<(CC * CC * 9 + 255) / 256, 256, 0, stream>>>(F(P_CLS_W1), w1c, CC * CC * 9);
    cvt_bf16_kernel<<<(2 * CC * 9 + 255) / 256, 256, 0, stream>>>(F(P_CLS_W2), w2c, 2 * CC * 9);

    auto gemmf = [&](const bf16_t* A, int lda, const bf16_t* Bt,
                     const float* bias, const float* res, int ldr,
                     float* Co, int ldc, int M, int Nc, int K, int act) {
        long tiles = (long)((M + 63) / 64) * (Nc / 16);
        int blocks = (int)((tiles + 3) / 4);
        gemm_bf16_wmma<<<blocks, 128, 0, stream>>>(A, lda, Bt, bias, res, ldr,
                                                   Co, nullptr, ldc, M, Nc, K, act);
    };
    auto gemmb = [&](const bf16_t* A, int lda, const bf16_t* Bt,
                     const float* bias, const float* res, int ldr,
                     bf16_t* Co, int ldc, int M, int Nc, int K, int act) {
        long tiles = (long)((M + 63) / 64) * (Nc / 16);
        int blocks = (int)((tiles + 3) / 4);
        gemm_bf16_wmma<<<blocks, 128, 0, stream>>>(A, lda, Bt, bias, res, ldr,
                                                   nullptr, Co, ldc, M, Nc, K, act);
    };
    auto lnorm = [&](const float* x, const float* g, const float* b, bf16_t* y, int M) {
        ln_kernel<<<(M + 255) / 256, 256, 0, stream>>>(x, g, b, y, M);
    };

    // ---------------- Phase A: graph scores, top-k, k-means ----------------
    {
        long nel = (long)RWS * CC;
        absdiff_rows_kernel<<<(int)((nel + 255) / 256), 256, 0, stream>>>(F(IN_X1), F(IN_X2), af);
    }
    edge_dot_kernel<<<(E + 255) / 256, 256, 0, stream>>>(af, ind0, vals, E);
    dis_kernel<<<(RWS + 255) / 256, 256, 0, stream>>>(vals, dis);
    score_kernel<<<(RWS + 255) / 256, 256, 0, stream>>>(af, F(P_GC_W), sg);
    fm_kernel<<<(RWS + 255) / 256, 256, 0, stream>>>(vals, ind0, dis, sg, F(P_GC_B), fmv, E);
    topk_rank_kernel<<<dim3(16, BB), 256, 0, stream>>>(fmv, idx);
    {
        long nel = (long)BB * KTOP * CC;
        gather_tok_kernel<<<(int)((nel + 255) / 256), 256, 0, stream>>>(F(IN_X1), idx, tg1);
        gather_tok_kernel<<<(int)((nel + 255) / 256), 256, 0, stream>>>(F(IN_X2), idx, tg2);
    }
    kmeans_kernel<<<BB, 256, 0, stream>>>(tg1, tok1);
    kmeans_kernel<<<BB, 256, 0, stream>>>(tg2, tok2);
    concat_pos_kernel<<<(BB * TT * CC + 255) / 256, 256, 0, stream>>>(tok1, tok2, F(P_POS), tokens);

    // ---------------- Phase B: token encoder + cross blocks ----------------
    lnorm(tokens, F(PE_LN1_G), F(PE_LN1_B), xn_tok, BB * TT);
    gemmf(xn_tok, CC, pe_qkv_t, nullptr, nullptr, 0, qkv_tok, 3 * INR,
          BB * TT, 3 * INR, CC, 0);
    attn_kernel<20><<<(BB * TT * 8 + 255) / 256, 256, 0, stream>>>(
        qkv_tok, 3 * INR, qkv_tok + INR, 3 * INR, qkv_tok + 2 * INR, 3 * INR,
        attn_tok, INR, TT, BB * TT);
    gemmf(attn_tok, INR, pe_out_t, F(PE_OUT_B), tokens, CC, tokens2, CC,
          BB * TT, CC, INR, 0);
    lnorm(tokens2, F(PE_LN2_G), F(PE_LN2_B), xn_tok, BB * TT);
    gemmb(xn_tok, CC, pe_ff1_t, F(PE_FF_B1), nullptr, 0, h_tok, MLPD,
          BB * TT, MLPD, CC, 1);
    gemmf(h_tok, MLPD, pe_ff2_t, F(PE_FF_B2), tokens2, CC, tokens2, CC,
          BB * TT, CC, MLPD, 0);
    split_tokens_kernel<<<(BB * TT * CC + 255) / 256, 256, 0, stream>>>(tokens2, t1e, t2e);

    auto cross = [&](const float* xin, const float* min_, float* outb) {
        int M = BB * CLUST;
        lnorm(xin, F(PC_LN1_G), F(PC_LN1_B), xn40, M);
        lnorm(min_, F(PC_LN1_G), F(PC_LN1_B), mn40, M);
        gemmf(xn40, CC, pc_q_t, nullptr, nullptr, 0, q40, INR, M, INR, CC, 0);
        gemmf(mn40, CC, pc_k_t, nullptr, nullptr, 0, k40, INR, M, INR, CC, 0);
        gemmf(mn40, CC, pc_v_t, nullptr, nullptr, 0, v40, INR, M, INR, CC, 0);
        attn_kernel<10><<<(M * 8 + 255) / 256, 256, 0, stream>>>(
            q40, INR, k40, INR, v40, INR, ao40, INR, CLUST, M);
        gemmf(ao40, INR, pc_out_t, F(PC_OUT_B), xin, CC, outb, CC, M, CC, INR, 0);
        lnorm(outb, F(PC_LN2_G), F(PC_LN2_B), xn40, M);
        gemmb(xn40, CC, pc_ff1_t, F(PC_FF_B1), nullptr, 0, h40, MLPD, M, MLPD, CC, 1);
        gemmf(h40, MLPD, pc_ff2_t, F(PC_FF_B2), outb, CC, outb, CC, M, CC, MLPD, 0);
    };
    cross(t1e, t2e, t1c);
    cross(t2e, t1e, t2c);

    // ---------------- Phase C: pixel decoders (heavy, WMMA) ----------------
    auto decoder = [&](const float* ximg, const float* mtok, float* outdec) {
        {
            long nel = (long)RWS * CC;
            nchw_to_rows_kernel<<<(int)((nel + 255) / 256), 256, 0, stream>>>(ximg, tcur);
        }
        for (int l = 0; l < DEPTH; ++l) {
            const bf16_t* qw = pd_q_t + (size_t)l * CC * INR;
            const bf16_t* kw = pd_k_t + (size_t)l * CC * INR;
            const bf16_t* vw = pd_v_t + (size_t)l * CC * INR;
            const bf16_t* ow = pd_out_t + (size_t)l * INR * CC;
            const bf16_t* w1 = pd_ff1_t + (size_t)l * CC * MLPD;
            const bf16_t* w2 = pd_ff2_t + (size_t)l * MLPD * CC;
            const float* ob = F(PD_OUT_B) + l * CC;
            const float* b1 = F(PD_FF_B1) + l * MLPD;
            const float* b2 = F(PD_FF_B2) + l * CC;
            const float* l1g = F(PD_LN1_G) + l * CC;
            const float* l1b = F(PD_LN1_B) + l * CC;
            const float* l2g = F(PD_LN2_G) + l * CC;
            const float* l2b = F(PD_LN2_B) + l * CC;
            lnorm(tcur, l1g, l1b, xnb, RWS);
            lnorm(mtok, l1g, l1b, mnb, BB * CLUST);
            gemmf(xnb, CC, qw, nullptr, nullptr, 0, qb, INR, RWS, INR, CC, 0);
            gemmf(mnb, CC, kw, nullptr, nullptr, 0, kb, INR, BB * CLUST, INR, CC, 0);
            gemmf(mnb, CC, vw, nullptr, nullptr, 0, vb, INR, BB * CLUST, INR, CC, 0);
            attn_kernel<10><<<(RWS * 8 + 255) / 256, 256, 0, stream>>>(
                qb, INR, kb, INR, vb, INR, aob, INR, NPIX, RWS);
            gemmf(aob, INR, ow, ob, tcur, CC, tcur, CC, RWS, CC, INR, 0);
            lnorm(tcur, l2g, l2b, xnb, RWS);
            gemmb(xnb, CC, w1, b1, nullptr, 0, hb, MLPD, RWS, MLPD, CC, 1);
            gemmf(hb, MLPD, w2, b2, tcur, CC, tcur, CC, RWS, CC, MLPD, 0);
        }
        hipMemcpyAsync(outdec, tcur, (size_t)RWS * CC * 4, hipMemcpyDeviceToDevice, stream);
    };
    decoder(F(IN_X1), t1c, dec1);
    decoder(F(IN_X2), t2c, dec2);

    // ---------------- Phase D: diff, upsample, classifier convs ------------
    {
        long nel = (long)RWS * CC;
        diff_to_nchw_kernel<<<(int)((nel + 255) / 256), 256, 0, stream>>>(dec1, dec2, diffn);
    }
    {
        long nel = (long)BB * CC * 256 * 256;
        upsample4_kernel<<<(int)((nel + 255) / 256), 256, 0, stream>>>(diffn, ups);
    }
    {
        long tiles = (long)((BB * 256 * 256 + 15) / 16) * ((CC + 15) / 16);
        conv3x3_wmma<<<(int)((tiles + 3) / 4), 128, 0, stream>>>(
            ups, w1c, nullptr, F(P_BN_G), F(P_BN_B), nullptr, c1o, BB, CC, CC, 256, 256, 1);
    }
    {
        long tiles = (long)((BB * 256 * 256 + 15) / 16) * 1;
        conv3x3_wmma<<<(int)((tiles + 3) / 4), 128, 0, stream>>>(
            c1o, w2c, F(P_CLS_B2), nullptr, nullptr, (float*)d_out, nullptr,
            BB, CC, 2, 256, 256, 0);
    }
}
